// LocalContrastiveLoss_25348896981163
// MI455X (gfx1250) — compile-verified, matrix operand order checked by reading, not run on hardware
//
#include <hip/hip_runtime.h>

// CDNA5 / gfx1250 wave32 WMMA implementation of the local contrastive loss.
// Pass 1: L2-normalize features and transpose (N,C,H,W) f32 -> (N,H,W,C) f16 in d_ws (64 MB).
// Pass 2: per 16-pixel row segment, compute all 121 neighborhood cosine logits with
//         v_wmma_f32_16x16x32_f16 (two 16x16 tiles per row-offset), accumulate per-slot
//         softmax denominators in registers (branchless masked FMAs), collapse to a
//         scalar via wave-reduce + atomicAdd.

typedef __attribute__((ext_vector_type(16))) _Float16 v16h;
typedef __attribute__((ext_vector_type(8)))  _Float16 v8h;
typedef __attribute__((ext_vector_type(8)))  float    v8f;

namespace {
constexpr int   NB = 4;      // batch
constexpr int   CC = 128;    // channels
constexpr int   HH = 256;
constexpr int   WW = 256;
constexpr int   NS = 5;      // neighborhood radius (K = 11)
constexpr float INV_TEMP = 10.0f;
constexpr float EPSV = 1e-6f;
}

__global__ void lcl_zero(float* out) {
    if (threadIdx.x == 0 && blockIdx.x == 0) out[0] = 0.0f;
}

// One block per (n, h) row. Tiled transpose through LDS so global reads
// (channel-major f32) and writes (pixel-major f16, 256B/pixel) are coalesced.
__global__ __launch_bounds__(256) void lcl_normalize(const float* __restrict__ feat,
                                                     _Float16* __restrict__ outp) {
    __shared__ float tile[CC][33];   // +1 pad: conflict-free column reads
    __shared__ float sscale[32];
    const int tid = threadIdx.x;
    const int n = blockIdx.x >> 8;
    const int h = blockIdx.x & 255;

    for (int w0 = 0; w0 < WW; w0 += 32) {
        // load 128 channels x 32 pixels (coalesced 128B rows)
        #pragma unroll
        for (int cc = 0; cc < CC; cc += 8) {
            const int c  = cc + (tid >> 5);
            const int wl = tid & 31;
            tile[c][wl] = feat[((((size_t)n * CC + c) * HH + h) * WW) + w0 + wl];
        }
        __syncthreads();
        // per-pixel inverse norm
        if (tid < 32) {
            float ss = 0.0f;
            #pragma unroll
            for (int c = 0; c < CC; ++c) { const float v = tile[c][tid]; ss += v * v; }
            sscale[tid] = 1.0f / fmaxf(sqrtf(ss), 1e-12f);
        }
        __syncthreads();
        // write pixel-major f16: 8 threads per pixel, 16 channels each (two b128 stores)
        {
            const int p = tid >> 3;   // pixel within chunk
            const int j = tid & 7;    // channel group: 16j..16j+15
            const float sc = sscale[p];
            v8h o0, o1;
            #pragma unroll
            for (int k = 0; k < 8; ++k) o0[k] = (_Float16)(tile[16 * j + k][p] * sc);
            #pragma unroll
            for (int k = 0; k < 8; ++k) o1[k] = (_Float16)(tile[16 * j + 8 + k][p] * sc);
            _Float16* dst = outp + (((((size_t)n * HH + h) * WW) + w0 + p) << 7) + 16 * j;
            *(v8h*)dst       = o0;
            *(v8h*)(dst + 8) = o1;
        }
        __syncthreads();
    }
}

__device__ inline v16h load_frag(const _Float16* p) {
    const v8h a = *(const v8h*)p;
    const v8h b = *(const v8h*)(p + 8);
    return __builtin_shufflevector(a, b, 0, 1, 2, 3, 4, 5, 6, 7, 8, 9, 10, 11, 12, 13, 14, 15);
}

// 8 waves per block; each wave owns one (n, h, w0..w0+15) segment.
__global__ __launch_bounds__(256) void lcl_main(const _Float16* __restrict__ fn,
                                                const int* __restrict__ labels,
                                                float* __restrict__ out) {
    const int lane = threadIdx.x & 31;
    const int wid  = blockIdx.x * (blockDim.x >> 5) + (threadIdx.x >> 5);  // [0, 16384)
    const int n  = wid >> 12;
    const int h  = (wid >> 4) & 255;
    const int w0 = (wid & 15) << 4;
    const int nL = lane & 15;
    const int hi = lane >> 4;

    // A fragments: 16 center pixels (M = nL), 128 channels, per-lane channel map
    // identical for A and B (contraction is permutation-invariant in k).
    const _Float16* ap = fn + (((((size_t)n * HH + h) * WW) + w0 + nL) << 7);
    v16h A[4];
    #pragma unroll
    for (int kc = 0; kc < 4; ++kc) A[kc] = load_frag(ap + kc * 32 + hi * 16);

    // center labels per C-element r (M = r + 8*hi)
    const size_t lrow = ((size_t)n * HH + h) * WW;
    int labc[8];
    #pragma unroll
    for (int r = 0; r < 8; ++r) labc[r] = labels[lrow + w0 + r + 8 * hi];

    // neighbor column for this lane in tile1 (s = -5) and tile2 (s = +11)
    const int col1 = w0 + nL - NS;
    const int col2 = w0 + nL + NS + 6;
    const bool cin1 = (col1 >= 0) && (col1 < WW);
    const bool cin2 = (col2 < WW);
    const int cc1 = min(max(col1, 0), WW - 1);
    const int cc2 = min(col2, WW - 1);

    // static per-slot masks (db-range x column-validity), as floats for branchless FMA
    float sm1[8], sm2[8];
    #pragma unroll
    for (int r = 0; r < 8; ++r) {
        const int M = r + 8 * hi;
        const int db1 = nL - NS - M;        // tile1: db = n - 5 - M
        const int db2 = nL + NS + 6 - M;    // tile2: db = n + 11 - M
        sm1[r] = (cin1 && db1 >= -NS && db1 <= NS) ? 1.0f : 0.0f;
        sm2[r] = (cin2 && db2 >= -NS && db2 <= NS) ? 1.0f : 0.0f;
    }

    float Dsum1[8] = {}, Sl1[8] = {}, Dsum2[8] = {}, Sl2[8] = {};
    int nvr = 0;

    for (int dy = -NS; dy <= NS; ++dy) {
        const int hr = h + dy;
        if ((unsigned)hr >= (unsigned)HH) continue;  // uniform per wave: EXEC stays all-ones
        ++nvr;

        const _Float16* b1p = fn + (((((size_t)n * HH + hr) * WW) + cc1) << 7);
        const _Float16* b2p = fn + (((((size_t)n * HH + hr) * WW) + cc2) << 7);
        v16h B1[4], B2[4];
        #pragma unroll
        for (int kc = 0; kc < 4; ++kc) {
            B1[kc] = load_frag(b1p + kc * 32 + hi * 16);
            B2[kc] = load_frag(b2p + kc * 32 + hi * 16);
        }
        const int lab1 = labels[((size_t)n * HH + hr) * WW + cc1];
        const int lab2 = labels[((size_t)n * HH + hr) * WW + cc2];

        v8f c1 = {}, c2 = {};
        #pragma unroll
        for (int kc = 0; kc < 4; ++kc) {
            c1 = __builtin_amdgcn_wmma_f32_16x16x32_f16(false, A[kc], false, B1[kc],
                                                        (short)0, c1, false, false);
            c2 = __builtin_amdgcn_wmma_f32_16x16x32_f16(false, A[kc], false, B2[kc],
                                                        (short)0, c2, false, false);
        }

        // D element r of this lane: M = r + 8*hi, neighbor column index nL.
        // Fixed (M, db) always lands in the same slot -> pure register accumulation.
        // Branchless: all values are bounded (|cos|<=1 -> exp(l)<=e^10.2), so compute
        // unconditionally and gate with mask FMAs (no EXEC manipulation between WMMAs).
        #pragma unroll
        for (int r = 0; r < 8; ++r) {
            const float m1 = (lab1 == labc[r]) ? sm1[r] : 0.0f;
            const float l1 = c1[r] * INV_TEMP;
            Dsum1[r] = fmaf(m1, __expf(l1), Dsum1[r]);
            Sl1[r]   = fmaf(m1, l1, Sl1[r]);

            const float m2 = (lab2 == labc[r]) ? sm2[r] : 0.0f;
            const float l2 = c2[r] * INV_TEMP;
            Dsum2[r] = fmaf(m2, __expf(l2), Dsum2[r]);
            Sl2[r]   = fmaf(m2, l2, Sl2[r]);
        }
    }

    // Per-slot finalize: sum over valid da of (log D_db - l) = nvr*log(D) - Sum(l).
    // log(EPS) when masked-out is finite and multiplied by 0.
    float local = 0.0f;
    const float fnvr = (float)nvr;
    #pragma unroll
    for (int r = 0; r < 8; ++r) {
        const int M = r + 8 * hi;
        const int w = w0 + M;
        const float nvc = (float)(min(NS, WW - 1 - w) + min(NS, w) + 1);
        const float inv_win = 1.0f / (fnvr * nvc);

        local += sm1[r] * (fnvr * __logf(Dsum1[r] + EPSV) - Sl1[r]) * inv_win;
        local += sm2[r] * (fnvr * __logf(Dsum2[r] + EPSV) - Sl2[r]) * inv_win;
    }
    local *= 1.0f / ((float)NB * (float)HH * (float)WW);

    // wave32 reduce, one atomic per wave
    #pragma unroll
    for (int off = 16; off > 0; off >>= 1) local += __shfl_down(local, off, 32);
    if (lane == 0) atomicAdd(out, local);
}

extern "C" void kernel_launch(void* const* d_in, const int* in_sizes, int n_in,
                              void* d_out, int out_size, void* d_ws, size_t ws_size,
                              hipStream_t stream) {
    (void)in_sizes; (void)n_in; (void)out_size; (void)ws_size;
    const float* feat  = (const float*)d_in[0];
    const int*   labs  = (const int*)d_in[1];
    float*       outp  = (float*)d_out;
    _Float16*    fnorm = (_Float16*)d_ws;   // needs NB*HH*WW*CC*2 = 64 MB scratch

    lcl_zero<<<1, 32, 0, stream>>>(outp);
    lcl_normalize<<<NB * HH, 256, 0, stream>>>(feat, fnorm);
    // 16384 segments, 8 waves (256 threads) per block -> 2048 blocks, no partial waves
    lcl_main<<<(NB * HH * (WW / 16)) / 8, 256, 0, stream>>>(fnorm, labs, outp);
}
